// ProbSparseAttention_8864812499638
// MI455X (gfx1250) — matrix operand order repaired and checked by reading, not compiled, not run on hardware
//
#include <hip/hip_runtime.h>
#include <hip/hip_bf16.h>

// ===================== CDNA5 / gfx1250 WMMA types =====================
typedef __attribute__((ext_vector_type(16))) __bf16 v16bf;
typedef __attribute__((ext_vector_type(8)))  float  v8f;
typedef __attribute__((ext_vector_type(4)))  unsigned int v4u;
typedef __attribute__((ext_vector_type(8)))  int v8i;
typedef __attribute__((ext_vector_type(4)))  int v4i;

struct U8 { uint4 a, b; };   // 32 bytes == v16bf

__device__ __forceinline__ unsigned short f2bf(float f) {
    unsigned int u = __float_as_uint(f);
    unsigned int r = u + 0x7FFFu + ((u >> 16) & 1u);  // RNE
    return (unsigned short)(r >> 16);
}

// ============ Tensor Data Mover: 2-D tile load (global -> LDS) ============
// D# per CDNA5 ISA: group0 = {count, lds_addr, global_addr, type=2},
// group1 = {data_size=2B, tensor_dim0/1, tile_dim0/1, tensor_dim0_stride}.
// Dims/strides in elements; dim0 is the contiguous dimension.
__device__ __forceinline__ void tdm_load_2d_bf16(unsigned lds_off, const void* gptr,
                                                 unsigned tile_d0, unsigned tile_d1,
                                                 unsigned tensor_d0, unsigned tensor_d1,
                                                 unsigned stride_elems) {
#if __has_builtin(__builtin_amdgcn_tensor_load_to_lds)
    unsigned long long ga = (unsigned long long)(size_t)gptr;
    v4u g0;
    g0.x = 1u;                                         // count=1 (valid user D#)
    g0.y = lds_off;                                    // lds_addr (bytes)
    g0.z = (unsigned)(ga & 0xFFFFFFFFu);               // global_addr[31:0]
    g0.w = (unsigned)((ga >> 32) & 0x01FFFFFFu) | 0x80000000u;  // addr[56:32] | type=2
    v8i g1;
    g1[0] = (int)(1u << 16);                           // wg_mask=0, data_size=1 (2 bytes)
    g1[1] = (int)((tensor_d0 & 0xFFFFu) << 16);        // tensor_dim0[15:0] @ bits 63:48
    g1[2] = (int)(((tensor_d0 >> 16) & 0xFFFFu) | ((tensor_d1 & 0xFFFFu) << 16));
    g1[3] = (int)(((tensor_d1 >> 16) & 0xFFFFu) | ((tile_d0 & 0xFFFFu) << 16));
    g1[4] = (int)(tile_d1 & 0xFFFFu);                  // tile_dim1; tile_dim2=0
    g1[5] = (int)stride_elems;                         // tensor_dim0_stride[31:0]
    g1[6] = 0;
    g1[7] = 0;
    v4i z4 = {0, 0, 0, 0};
#if __clang_major__ >= 23
    v8i z8 = {0, 0, 0, 0, 0, 0, 0, 0};
    __builtin_amdgcn_tensor_load_to_lds(g0, g1, z4, z4, z8, 0);
#else
    __builtin_amdgcn_tensor_load_to_lds(g0, g1, z4, z4, 0);
#endif
#else
    (void)lds_off; (void)gptr; (void)tile_d0; (void)tile_d1;
    (void)tensor_d0; (void)tensor_d1; (void)stride_elems;
#endif
}

// ===================== fp32 -> bf16 convert (row major) =====================
__global__ void f32_to_bf16_kernel(const float* __restrict__ in,
                                   unsigned short* __restrict__ out, int n) {
    int i = blockIdx.x * blockDim.x + threadIdx.x;
    if (i < n) out[i] = f2bf(in[i]);
}

// ============ fp32 [R][C] -> bf16 transposed [C][R] (weights only, one-shot) ============
__global__ void f32_to_bf16_transpose_kernel(const float* __restrict__ in,
                                             unsigned short* __restrict__ out,
                                             int R, int C) {
    int i = blockIdx.x * blockDim.x + threadIdx.x;
    if (i < R * C) {
        int r = i / C, c = i % C;
        out[(size_t)c * R + r] = f2bf(in[(size_t)r * C + c]);
    }
}

// ===================== bf16 WMMA GEMM, double-buffered TDM pipeline =====================
// C[M,N] = A[M,K] * Bt[N,K]^T + bias.  Block tile 128x64, BK=32, 4 waves, 2 LDS buffers.
// Wave 0 issues TDM for tile k+1 into the ping-pong buffer, then waits tensorcnt<=2
// (TDM ops retire in order per wave, so the current tile's 2 ops are then complete)
// while tile k+1 streams in behind the compute.
__global__ __launch_bounds__(128)
void gemm_bf16_tdm_kernel(const unsigned short* __restrict__ A,   // [M,K] bf16
                          const unsigned short* __restrict__ Bt,  // [N,K] bf16 (pre-transposed)
                          const float* __restrict__ bias,
                          float* __restrict__ outF,
                          unsigned short* __restrict__ outB,
                          int M, int N, int K) {
    __shared__ __align__(16) unsigned short Ash[2][128][32];  // 16 KB
    __shared__ __align__(16) unsigned short Bsh[2][64][32];   //  8 KB

    const int t    = threadIdx.x;
    const int lane = t & 31;
    const int w    = t >> 5;
    const int m0   = blockIdx.y * 128;
    const int n0   = blockIdx.x * 64;
    const int mloc = lane & 15;
    const int half = lane >> 4;
    const int NK   = K / 32;

    v8f c[2][4];
    #pragma unroll
    for (int mt = 0; mt < 2; ++mt)
        #pragma unroll
        for (int nt = 0; nt < 4; ++nt) c[mt][nt] = v8f{};

    // prologue: DMA tile 0 into buffer 0
    if (t < 32) {
        tdm_load_2d_bf16((unsigned)(size_t)&Ash[0][0][0], A + (size_t)m0 * K,
                         32, 128, (unsigned)K, (unsigned)M, (unsigned)K);
        tdm_load_2d_bf16((unsigned)(size_t)&Bsh[0][0][0], Bt + (size_t)n0 * K,
                         32, 64, (unsigned)K, (unsigned)N, (unsigned)K);
    }

    for (int kt = 0; kt < NK; ++kt) {
        const int buf = kt & 1;
        if (t < 32) {
            if (kt + 1 < NK) {
                const int nb = (kt + 1) & 1;
                const int k1 = (kt + 1) * 32;
                tdm_load_2d_bf16((unsigned)(size_t)&Ash[nb][0][0], A + (size_t)m0 * K + k1,
                                 32, 128, (unsigned)K, (unsigned)M, (unsigned)K);
                tdm_load_2d_bf16((unsigned)(size_t)&Bsh[nb][0][0], Bt + (size_t)n0 * K + k1,
                                 32, 64, (unsigned)K, (unsigned)N, (unsigned)K);
                __builtin_amdgcn_s_wait_tensorcnt(2);  // current tile's 2 ops retired
            } else {
                __builtin_amdgcn_s_wait_tensorcnt(0);
            }
        }
        __syncthreads();

        U8 au0, au1;
        au0.a = *reinterpret_cast<const uint4*>(&Ash[buf][w * 32 + mloc][half * 8]);
        au0.b = *reinterpret_cast<const uint4*>(&Ash[buf][w * 32 + mloc][16 + half * 8]);
        au1.a = *reinterpret_cast<const uint4*>(&Ash[buf][w * 32 + 16 + mloc][half * 8]);
        au1.b = *reinterpret_cast<const uint4*>(&Ash[buf][w * 32 + 16 + mloc][16 + half * 8]);
        v16bf a0 = __builtin_bit_cast(v16bf, au0);
        v16bf a1 = __builtin_bit_cast(v16bf, au1);

        #pragma unroll
        for (int nt = 0; nt < 4; ++nt) {
            U8 bu;
            bu.a = *reinterpret_cast<const uint4*>(&Bsh[buf][nt * 16 + mloc][half * 8]);
            bu.b = *reinterpret_cast<const uint4*>(&Bsh[buf][nt * 16 + mloc][16 + half * 8]);
            v16bf bv = __builtin_bit_cast(v16bf, bu);
            c[0][nt] = __builtin_amdgcn_wmma_f32_16x16x32_bf16(
                false, a0, false, bv, (short)0, c[0][nt], false, false);
            c[1][nt] = __builtin_amdgcn_wmma_f32_16x16x32_bf16(
                false, a1, false, bv, (short)0, c[1][nt], false, false);
        }
        __syncthreads();  // buffer `buf` may be re-issued at the top of iteration kt+1
    }

    // ---- epilogue: bias + store (C/D layout: n = L%16, m = i + 8*half) ----
    #pragma unroll
    for (int mt = 0; mt < 2; ++mt) {
        #pragma unroll
        for (int nt = 0; nt < 4; ++nt) {
            int n = n0 + nt * 16 + mloc;
            float bval = bias ? bias[n] : 0.0f;
            #pragma unroll
            for (int i = 0; i < 8; ++i) {
                int m = m0 + w * 32 + mt * 16 + i + 8 * half;
                float val = c[mt][nt][i] + bval;
                size_t idx = (size_t)m * N + n;
                if (outF) outF[idx] = val;
                if (outB) outB[idx] = f2bf(val);
            }
        }
    }
}

// ===================== per-column sum of V (sparse-softmax closed form) =====================
__global__ void zero_f32_kernel(float* __restrict__ p, int n) {
    int i = blockIdx.x * blockDim.x + threadIdx.x;
    if (i < n) p[i] = 0.0f;
}

__global__ void vsum_kernel(const float* __restrict__ V, float* __restrict__ Vsum) {
    // grid (8 col-blocks x 8 row-blocks), 128 threads: 256-row partial sums + f32 atomic
    int d  = blockIdx.x * 128 + threadIdx.x;
    int r0 = blockIdx.y * 256;
    float s = 0.0f;
    for (int q = r0; q < r0 + 256; ++q) s += V[(size_t)q * 1024 + d];
    atomicAdd(&Vsum[d], s);
}

// ===================== ProbSparse attention core =====================
// One wave per (16-query tile, head). Streams 128 key tiles; each scored with
// 2 chained v_wmma_f32_16x16x32_bf16 (K=64); per-row top-16 kept in LDS; then
// closed-form sparse softmax: out = sum_i (p_i - p0) V[idx_i] + p0 * colsum(V).
// The next key tile's B operands are preloaded into registers so the global
// loads overlap the LDS top-k scan.
__global__ __launch_bounds__(32)
void sparse_attn_kernel(const unsigned short* __restrict__ Qb,   // [2048,1024] bf16
                        const unsigned short* __restrict__ Kb,   // [2048,1024] bf16
                        const float* __restrict__ V,             // [2048,1024] f32
                        const float* __restrict__ Vsum,          // [1024] f32
                        unsigned short* __restrict__ ctxb) {     // [2048,1024] bf16
    __shared__ float sS[16][16];
    __shared__ float topv[16][16];
    __shared__ int   topi[16][16];
    __shared__ float pz[16];

    const int lane = threadIdx.x;
    const int mloc = lane & 15;
    const int half = lane >> 4;
    const int q0   = blockIdx.x * 16;
    const int hb   = blockIdx.y * 64;

    // ---- Q tile once: two A operands (K=0..31, K=32..63) ----
    const uint4* qp = reinterpret_cast<const uint4*>(Qb + (size_t)(q0 + mloc) * 1024 + hb);
    U8 a0u, a1u;
    a0u.a = qp[half];
    a0u.b = qp[2 + half];
    a1u.a = qp[4 + half];
    a1u.b = qp[6 + half];
    v16bf aA = __builtin_bit_cast(v16bf, a0u);
    v16bf aB = __builtin_bit_cast(v16bf, a1u);

    float minv = -3.0e38f;
    int   minslot = 0;
    if (lane < 16) {
        #pragma unroll
        for (int i = 0; i < 16; ++i) { topv[lane][i] = -3.0e38f; topi[lane][i] = 0; }
    }
    __syncthreads();

    // preload key tile 0's B operands
    U8 b0u, b1u;
    {
        const uint4* kp = reinterpret_cast<const uint4*>(Kb + (size_t)mloc * 1024 + hb);
        b0u.a = kp[half];
        b0u.b = kp[2 + half];
        b1u.a = kp[4 + half];
        b1u.b = kp[6 + half];
    }

    for (int kt = 0; kt < 128; ++kt) {
        U8 n0u = b0u, n1u = b1u;
        if (kt + 1 < 128) {  // preload next tile; in flight during the top-k scan
            const uint4* kp = reinterpret_cast<const uint4*>(
                Kb + (size_t)((kt + 1) * 16 + mloc) * 1024 + hb);
            n0u.a = kp[half];
            n0u.b = kp[2 + half];
            n1u.a = kp[4 + half];
            n1u.b = kp[6 + half];
        }

        v16bf bA = __builtin_bit_cast(v16bf, b0u);
        v16bf bB = __builtin_bit_cast(v16bf, b1u);
        v8f c = {};
        c = __builtin_amdgcn_wmma_f32_16x16x32_bf16(false, aA, false, bA, (short)0, c, false, false);
        c = __builtin_amdgcn_wmma_f32_16x16x32_bf16(false, aB, false, bB, (short)0, c, false, false);

        #pragma unroll
        for (int i = 0; i < 8; ++i)
            sS[i + 8 * half][mloc] = c[i] * 0.125f;   // 1/sqrt(64)
        __syncthreads();

        if (lane < 16) {
            #pragma unroll
            for (int cc = 0; cc < 16; ++cc) {
                float v = sS[lane][cc];
                if (v > minv) {
                    topv[lane][minslot] = v;
                    topi[lane][minslot] = kt * 16 + cc;
                    minv = topv[lane][0]; minslot = 0;
                    #pragma unroll
                    for (int j = 1; j < 16; ++j) {
                        float tv = topv[lane][j];
                        if (tv < minv) { minv = tv; minslot = j; }
                    }
                }
            }
        }
        __syncthreads();

        b0u = n0u;
        b1u = n1u;
    }

    // ---- sparse softmax: the (S-16) zero entries participate ----
    if (lane < 16) {
        float mx = 0.0f;
        #pragma unroll
        for (int i = 0; i < 16; ++i) mx = fmaxf(mx, topv[lane][i]);
        float e0 = __expf(-mx);
        float Z = 2032.0f * e0;                 // (2048-16) zeros
        #pragma unroll
        for (int i = 0; i < 16; ++i) {
            float e = __expf(topv[lane][i] - mx);
            Z += e;
            topv[lane][i] = e;
        }
        float invZ = 1.0f / Z;
        #pragma unroll
        for (int i = 0; i < 16; ++i) topv[lane][i] *= invZ;
        pz[lane] = e0 * invZ;
    }
    __syncthreads();

    const int d0 = lane * 2;
    for (int r = 0; r < 16; ++r) {
        float p0 = pz[r];
        float acc0 = p0 * Vsum[hb + d0];
        float acc1 = p0 * Vsum[hb + d0 + 1];
        #pragma unroll
        for (int i = 0; i < 16; ++i) {
            float wgt = topv[r][i] - p0;
            int idx = topi[r][i];
            const float2 vv = *reinterpret_cast<const float2*>(
                V + (size_t)idx * 1024 + hb + d0);
            acc0 += wgt * vv.x;
            acc1 += wgt * vv.y;
        }
        size_t ob = (size_t)(q0 + r) * 1024 + hb + d0;
        ctxb[ob]     = f2bf(acc0);
        ctxb[ob + 1] = f2bf(acc1);
    }
}

// ===================== host-side launch =====================
extern "C" void kernel_launch(void* const* d_in, const int* in_sizes, int n_in,
                              void* d_out, int out_size, void* d_ws, size_t ws_size,
                              hipStream_t stream) {
    const float* x  = (const float*)d_in[0];
    const float* Wq = (const float*)d_in[1];
    const float* bq = (const float*)d_in[2];
    const float* Wk = (const float*)d_in[3];
    const float* bk = (const float*)d_in[4];
    const float* Wv = (const float*)d_in[5];
    const float* bv = (const float*)d_in[6];
    const float* Wo = (const float*)d_in[7];
    const float* bo = (const float*)d_in[8];

    char* ws = (char*)d_ws;
    const size_t MB = 1u << 20;
    unsigned short* xb   = (unsigned short*)(ws + 0 * MB);      // 4 MB  x, bf16 [2048,1024]
    unsigned short* wqt  = (unsigned short*)(ws + 4 * MB);      // 2 MB  Wq^T bf16 [N,K]
    unsigned short* wkt  = (unsigned short*)(ws + 6 * MB);      // 2 MB
    unsigned short* wvt  = (unsigned short*)(ws + 8 * MB);      // 2 MB
    unsigned short* wot  = (unsigned short*)(ws + 10 * MB);     // 2 MB
    unsigned short* qb   = (unsigned short*)(ws + 12 * MB);     // 4 MB  Q bf16
    unsigned short* kb   = (unsigned short*)(ws + 16 * MB);     // 4 MB  K bf16
    float*          vf   = (float*)(ws + 20 * MB);              // 8 MB  V f32
    float*          vsum = (float*)(ws + 28 * MB);              // 4 KB
    unsigned short* ctxb = (unsigned short*)(ws + 28 * MB + 65536); // 4 MB

    const int SD = 2048 * 1024;
    const int DD = 1024 * 1024;

    f32_to_bf16_kernel<<<SD / 256, 256, 0, stream>>>(x, xb, SD);
    f32_to_bf16_transpose_kernel<<<DD / 256, 256, 0, stream>>>(Wq, wqt, 1024, 1024);
    f32_to_bf16_transpose_kernel<<<DD / 256, 256, 0, stream>>>(Wk, wkt, 1024, 1024);
    f32_to_bf16_transpose_kernel<<<DD / 256, 256, 0, stream>>>(Wv, wvt, 1024, 1024);
    f32_to_bf16_transpose_kernel<<<DD / 256, 256, 0, stream>>>(Wo, wot, 1024, 1024);

    dim3 gg(1024 / 64, 2048 / 128);  // (16, 16) block tiles of 128x64
    gemm_bf16_tdm_kernel<<<gg, 128, 0, stream>>>(xb, wqt, bq, nullptr, qb, 2048, 1024, 1024);
    gemm_bf16_tdm_kernel<<<gg, 128, 0, stream>>>(xb, wkt, bk, nullptr, kb, 2048, 1024, 1024);
    gemm_bf16_tdm_kernel<<<gg, 128, 0, stream>>>(xb, wvt, bv, vf,      nullptr, 2048, 1024, 1024);

    zero_f32_kernel<<<8, 128, 0, stream>>>(vsum, 1024);
    dim3 gv(8, 8);
    vsum_kernel<<<gv, 128, 0, stream>>>(vf, vsum);

    dim3 ga(128, 16);  // (query tiles, heads)
    sparse_attn_kernel<<<ga, 32, 0, stream>>>(qb, kb, vf, vsum, ctxb);

    gemm_bf16_tdm_kernel<<<gg, 128, 0, stream>>>(ctxb, wot, bo, (float*)d_out, nullptr,
                                                 2048, 1024, 1024);
}
